// SkipGramModel_36086315221340
// MI455X (gfx1250) — compile-verified, matrix operand order checked by reading, not run on hardware
//
#include <hip/hip_runtime.h>
#include <cstdint>
#include <cstddef>

// ---------------------------------------------------------------------------
// SkipGram negative-sampling loss for MI455X (gfx1250, wave32, WMMA).
// Memory-bound (~335 MB compulsory fp32 reads -> ~14.4 us @ 23.3 TB/s).
// bf16 WMMA (V_WMMA_F32_16X16X32_BF16) keeps matrix math far below the HBM
// floor. GEMM2 stages B tiles with the Tensor Data Mover (TENSOR_LOAD_TO_LDS,
// LDS padding + K-tail zero-fill done by the D#) and A tiles with
// GLOBAL_LOAD_ASYNC_TO_LDS (ASYNCcnt).
// ---------------------------------------------------------------------------

typedef __attribute__((ext_vector_type(16))) __bf16        v16bf;
typedef __attribute__((ext_vector_type(8)))  float         v8f;
typedef __attribute__((ext_vector_type(8)))  unsigned int  v8u;
typedef __attribute__((ext_vector_type(4)))  unsigned int  v4u;
typedef __attribute__((ext_vector_type(8)))  int           v8i;
typedef __attribute__((ext_vector_type(4)))  int           v4i;

#define D_EMB 512
#define P_POS 128
#define N_NEG 512

__device__ __forceinline__ unsigned short f32_bf16(float f) {
  unsigned int u = __builtin_bit_cast(unsigned int, f);
  u += 0x7FFFu + ((u >> 16) & 1u);          // round-to-nearest-even
  return (unsigned short)(u >> 16);
}

__device__ __forceinline__ unsigned int pk_bf16(float lo, float hi) {
#if __has_builtin(__builtin_amdgcn_cvt_pk_bf16_f32)
  typedef __attribute__((ext_vector_type(2))) __bf16 bf16x2;
  bf16x2 v = __builtin_amdgcn_cvt_pk_bf16_f32(lo, hi);
  return __builtin_bit_cast(unsigned int, v);
#else
  return ((unsigned int)f32_bf16(hi) << 16) | (unsigned int)f32_bf16(lo);
#endif
}

__device__ __forceinline__ void wait_async0() {
#if __has_builtin(__builtin_amdgcn_s_wait_asynccnt)
  __builtin_amdgcn_s_wait_asynccnt(0);
#else
  asm volatile("s_wait_asynccnt 0" ::: "memory");
#endif
}

// Per-lane async global->LDS copies. ldsoff = LDS byte offset (low 32 bits of
// the generic shared-space address per the aperture rules).
__device__ __forceinline__ void async_g2l_b128(unsigned ldsoff,
                                               const float* gaddr) {
  asm volatile("global_load_async_to_lds_b128 %0, %1, off"
               :: "v"(ldsoff), "v"(gaddr) : "memory");
}
__device__ __forceinline__ void async_g2l_b32(unsigned ldsoff,
                                              const float* gaddr) {
  asm volatile("global_load_async_to_lds_b32 %0, %1, off"
               :: "v"(ldsoff), "v"(gaddr) : "memory");
}

__global__ void zero_k(float* p, int n) {
  int i = blockIdx.x * 256 + threadIdx.x;
  if (i < n) p[i] = 0.0f;
}

// ---------------------------------------------------------------------------
// GEMM1: C1[144,512] += A1[144,Nv] * Ein^T   (split-K, bf16 WMMA)
//   A1 row 0 = center_word, rows 1..128 = positive_words, rows 129..143 = 0.
//   B[t,j] = Ein[j*Nv + t]  (column of Ein, contiguous along K; odd row
//   stride 50257 forbids wide aligned loads -> direct dword gathers).
// ---------------------------------------------------------------------------
__global__ __launch_bounds__(256) void gemm_pos(
    const float* __restrict__ center, const float* __restrict__ posw,
    const float* __restrict__ Ein, float* __restrict__ C1, int Nv, int kslab)
{
  __shared__ unsigned int a_sh[16][20];     // packed bf16 pairs, K-pair index

  const int tid   = threadIdx.x;
  const int lane  = tid & 31;
  const int wave  = tid >> 5;
  const int khalf = lane >> 4;              // 0: K 0..7/16..23, 1: +8
  const int jn    = lane & 15;
  const int m_base = blockIdx.y * 16;
  const int j_off  = blockIdx.z * 128 + wave * 16;

  const int k_begin = blockIdx.x * kslab;
  const int k_end   = min(k_begin + kslab, Nv);

  const int arow = m_base + (tid >> 4);
  const float* rp;
  if (arow == 0)          rp = center;
  else if (arow <= P_POS) rp = posw + (size_t)(arow - 1) * Nv;
  else                    rp = nullptr;
  const int akk = (tid & 15) * 2;

  const float* bbase = Ein + (size_t)(j_off + jn) * Nv;

  v8f c = {};

  for (int k0 = k_begin; k0 < k_end; k0 += 32) {
    const bool full = (k0 + 32) <= Nv;

    __syncthreads();
    {   // stage A tile (16 x 32) as bf16 pairs
      int t = k0 + akk;
      float f0 = 0.f, f1 = 0.f;
      if (rp) {
        if (full)            { f0 = rp[t]; f1 = rp[t + 1]; }
        else { if (t < Nv)     f0 = rp[t];
               if (t + 1 < Nv) f1 = rp[t + 1]; }
      }
      a_sh[tid >> 4][tid & 15] = pk_bf16(f0, f1);
    }
    __syncthreads();

    if (k0 + 256 < k_end) __builtin_prefetch(bbase + k0 + 256, 0, 1);

    v8u au;
#pragma unroll
    for (int r = 0; r < 8; ++r) {
      int k = ((r & 4) << 2) + khalf * 8 + (r & 3) * 2;
      au[r] = a_sh[jn][k >> 1];
    }

    v8u bu;
#pragma unroll
    for (int r = 0; r < 8; ++r) {
      int k = k0 + ((r & 4) << 2) + khalf * 8 + (r & 3) * 2;
      float f0, f1;
      if (full) { f0 = bbase[k]; f1 = bbase[k + 1]; }
      else {
        f0 = (k     < Nv) ? bbase[k]     : 0.f;
        f1 = (k + 1 < Nv) ? bbase[k + 1] : 0.f;
      }
      bu[r] = pk_bf16(f0, f1);
    }

    v16bf a = __builtin_bit_cast(v16bf, au);
    v16bf b = __builtin_bit_cast(v16bf, bu);
    c = __builtin_amdgcn_wmma_f32_16x16x32_bf16(false, a, false, b,
                                                (short)0, c, false, false);
  }

#pragma unroll
  for (int i = 0; i < 8; ++i) {
    int m = m_base + khalf * 8 + i;
    atomicAdd(&C1[(size_t)m * D_EMB + j_off + jn], c[i]);
  }
}

// ---------------------------------------------------------------------------
// GEMM2: C2[512,512] += negw[512,Nv] * Eout[Nv,512]   (split-K, bf16 WMMA)
//   B tile 32(k) x 128(j): Tensor Data Mover DMA into LDS; D# pad fields
//   give row stride 132 floats (bank-conflict-free column gathers) and
//   tensor_dim1 = Nv - k0 zero-fills the ragged K tail.
//   A tile: per-lane GLOBAL_LOAD_ASYNC_TO_LDS_B32 (odd global row stride).
// ---------------------------------------------------------------------------
__global__ __launch_bounds__(256) void gemm_neg(
    const float* __restrict__ negw, const float* __restrict__ Eout,
    float* __restrict__ C2, int Nv, int kslab)
{
  __shared__ __align__(16) float a_f32[16][36];    // [m][k], padded stride
  __shared__ __align__(16) float b_f32[32][132];   // [k][j], padded stride

  const int tid   = threadIdx.x;
  const int lane  = tid & 31;
  const int wave  = tid >> 5;
  const int khalf = lane >> 4;
  const int jn    = lane & 15;
  const int m_base = blockIdx.y * 16;
  const int jb     = blockIdx.z * 128;
  const int j_off  = jb + wave * 16;
  const int jl     = wave * 16 + jn;

  const int k_begin = blockIdx.x * kslab;
  const int k_end   = min(k_begin + kslab, Nv);

  const float* rp = negw + (size_t)(m_base + (tid >> 4)) * Nv;
  const int akk = (tid & 15) * 2;
  // scalar (wave-uniform) predicate so only wave 0 executes the TDM op
  const bool wave0 = ((__builtin_amdgcn_readfirstlane(tid) >> 5) == 0);

  v8f c = {};

  for (int k0 = k_begin; k0 < k_end; k0 += 32) {
    const bool full = (k0 + 32) <= Nv;

    __syncthreads();

    // ---- stage A tile (fp32) ----
    if (full) {
      unsigned la = (unsigned)(uintptr_t)&a_f32[tid >> 4][akk];
      async_g2l_b32(la,     rp + k0 + akk);
      async_g2l_b32(la + 4, rp + k0 + akk + 1);
    } else {
      int t = k0 + akk;
      a_f32[tid >> 4][akk]     = (t     < Nv) ? rp[t]     : 0.f;
      a_f32[tid >> 4][akk + 1] = (t + 1 < Nv) ? rp[t + 1] : 0.f;
    }

    // ---- stage B tile (fp32) ----
#if __has_builtin(__builtin_amdgcn_tensor_load_to_lds)
    if (wave0) {
      unsigned long long ga =
          (unsigned long long)(uintptr_t)(Eout + (size_t)k0 * D_EMB + jb);
      unsigned ldsb = (unsigned)(uintptr_t)&b_f32[0][0];
      unsigned rem  = (unsigned)(Nv - k0);     // remaining K rows (> 0)

      // D# group 0: count=1 | lds_addr | global_addr[56:0] | type=2
      v4u g0 = { 1u, ldsb,
                 (unsigned)(ga & 0xFFFFFFFFull),
                 (unsigned)((ga >> 32) & 0x01FFFFFFull) | (2u << 30) };
      // D# group 1: data_size=4B, pad after 128 DW by 4 DW, tensor 512 x rem,
      //             tile 128 x 32, dim0 stride 512
      v8i g1;
      g1[0] = (int)((2u << 16) | (1u << 20) | (6u << 22) | (3u << 25));
      g1[1] = (int)(((unsigned)D_EMB & 0xFFFFu) << 16);        // td0 lo
      g1[2] = (int)((rem & 0xFFFFu) << 16);                    // td0 hi | td1 lo
      g1[3] = (int)((128u << 16) | (rem >> 16));               // tile0 | td1 hi
      g1[4] = 32;                                              // tile1 (tile2=0)
      g1[5] = D_EMB;                                           // stride0 lo
      g1[6] = 0;                                               // stride0 hi | stride1 lo
      g1[7] = 0;
      v4i g2 = {0, 0, 0, 0};
      v4i g3 = {0, 0, 0, 0};
      v8i g4 = {0, 0, 0, 0, 0, 0, 0, 0};   // extra group (6-arg builtin form)
      __builtin_amdgcn_tensor_load_to_lds(g0, g1, g2, g3, g4, 0);
      __builtin_amdgcn_s_wait_tensorcnt(0);
    }
#else
    if (full) {
#pragma unroll
      for (int i = 0; i < 4; ++i) {
        int u    = tid + i * 256;
        int krow = u >> 5;
        int c4   = (u & 31) * 4;
        const float* gsrc = Eout + (size_t)(k0 + krow) * D_EMB + jb + c4;
        async_g2l_b128((unsigned)(uintptr_t)&b_f32[krow][c4], gsrc);
      }
    } else {
#pragma unroll
      for (int i = 0; i < 4; ++i) {
        int u    = tid + i * 256;
        int krow = u >> 5;
        int c4   = (u & 31) * 4;
        int t    = k0 + krow;
        float4 f = {0.f, 0.f, 0.f, 0.f};
        if (t < Nv) f = *(const float4*)(Eout + (size_t)t * D_EMB + jb + c4);
        b_f32[krow][c4 + 0] = f.x;
        b_f32[krow][c4 + 1] = f.y;
        b_f32[krow][c4 + 2] = f.z;
        b_f32[krow][c4 + 3] = f.w;
      }
    }
#endif

    wait_async0();          // own wave's async A (and B) copies landed
    __syncthreads();        // all waves' staging visible

    v8u au, bu;
#pragma unroll
    for (int r = 0; r < 8; ++r) {
      int k = ((r & 4) << 2) + khalf * 8 + (r & 3) * 2;
      au[r] = pk_bf16(a_f32[jn][k], a_f32[jn][k + 1]);
      bu[r] = pk_bf16(b_f32[k][jl], b_f32[k + 1][jl]);
    }

    v16bf a = __builtin_bit_cast(v16bf, au);
    v16bf b = __builtin_bit_cast(v16bf, bu);
    c = __builtin_amdgcn_wmma_f32_16x16x32_bf16(false, a, false, b,
                                                (short)0, c, false, false);
  }

#pragma unroll
  for (int i = 0; i < 8; ++i) {
    int m = m_base + khalf * 8 + i;
    atomicAdd(&C2[(size_t)m * D_EMB + j_off + jn], c[i]);
  }
}

// ---------------------------------------------------------------------------
// Scores + stable log-sigmoid + scalar reduction. One wave per row.
// ---------------------------------------------------------------------------
__global__ __launch_bounds__(256) void score_k(
    const float* __restrict__ C1, const float* __restrict__ C2,
    float* __restrict__ out)
{
  const int lane = threadIdx.x & 31;
  const int wid  = blockIdx.x * 8 + (threadIdx.x >> 5);

  const float* v = C1;                       // row 0 = input_embedding
  const float* row;
  float sgn;
  if (wid < P_POS) { row = C1 + (size_t)(1 + wid) * D_EMB;       sgn =  1.f; }
  else             { row = C2 + (size_t)(wid - P_POS) * D_EMB;   sgn = -1.f; }

  float s = 0.f;
#pragma unroll 4
  for (int j = lane; j < D_EMB; j += 32) s += v[j] * row[j];
#pragma unroll
  for (int off = 16; off; off >>= 1) s += __shfl_xor(s, off, 32);

  if (lane == 0) {
    float x  = sgn * s;
    float ls = fminf(x, 0.f) - log1pf(expf(-fabsf(x)));
    atomicAdd(out, -ls);
  }
}

// ---------------------------------------------------------------------------
extern "C" void kernel_launch(void* const* d_in, const int* in_sizes, int n_in,
                              void* d_out, int out_size, void* d_ws,
                              size_t ws_size, hipStream_t stream) {
  (void)n_in; (void)out_size; (void)ws_size;

  const float* center = (const float*)d_in[0];
  const float* posw   = (const float*)d_in[1];
  const float* negw   = (const float*)d_in[2];
  const float* Ein    = (const float*)d_in[3];   // [D, N]
  const float* Eout   = (const float*)d_in[4];   // [N, D]
  float* out = (float*)d_out;
  const int Nv = in_sizes[0];

  float* C1 = (float*)d_ws;                      // [144, 512]
  float* C2 = C1 + 144 * D_EMB;                  // [512, 512]

  const int zc = (144 + 512) * D_EMB;
  zero_k<<<dim3((zc + 255) / 256), 256, 0, stream>>>(C1, zc);
  zero_k<<<dim3(1), 256, 0, stream>>>(out, 1);

  const int ks1 = 28;
  int kslab1 = (((Nv + ks1 - 1) / ks1) + 31) & ~31;
  int nsl1   = (Nv + kslab1 - 1) / kslab1;
  gemm_pos<<<dim3(nsl1, 9, 4), 256, 0, stream>>>(center, posw, Ein, C1, Nv,
                                                 kslab1);

  const int ks2 = 12;
  int kslab2 = (((Nv + ks2 - 1) / ks2) + 31) & ~31;
  int nsl2   = (Nv + kslab2 - 1) / kslab2;
  gemm_neg<<<dim3(nsl2, 32, 4), 256, 0, stream>>>(negw, Eout, C2, Nv, kslab2);

  score_k<<<dim3((P_POS + N_NEG) / 8), 256, 0, stream>>>(C1, C2, out);
}
